// CrossAttentionBlock_1726576853276
// MI455X (gfx1250) — matrix operand order repaired and checked by reading, not compiled, not run on hardware
//
#include <hip/hip_runtime.h>
#include <hip/hip_bf16.h>

typedef __attribute__((ext_vector_type(16))) _Float16 v16h;
typedef __attribute__((ext_vector_type(8)))  _Float16 v8h;
typedef __attribute__((ext_vector_type(8)))  float    v8f;

#define WMMA_F16(a, b, c) \
  __builtin_amdgcn_wmma_f32_16x16x32_f16(false, (a), false, (b), (short)0, (c), false, false)

// ---- fragment loaders (wave32, v_wmma_f32_16x16x32_f16 layouts) ------------
// A (16x32 f16): lane m=lane&15; K-halves: [kh..kh+7] then [16+kh..16+kh+7],
// kh = (lane>>4)*8.  Source row-major [m][k], row stride ld (halves).
__device__ __forceinline__ v16h frag_a(const _Float16* p, int ld, int lane) {
  const _Float16* r = p + (lane & 15) * ld + ((lane >> 4) << 3);
  v8h lo = *(const v8h*)(r);
  v8h hi = *(const v8h*)(r + 16);
  v16h a;
#pragma unroll
  for (int i = 0; i < 8; ++i) { a[i] = lo[i]; a[8 + i] = hi[i]; }
  return a;
}

// B (32x16 f16): lane n=lane&15, K = [kb..kb+7] then [16+kb..16+kb+7], kb=(lane>>4)*8.
// Source holds the TRANSPOSED operand row-major: src[n][k], stride ld.
__device__ __forceinline__ v16h frag_bt(const _Float16* p, int ld, int lane) {
  const _Float16* r = p + (lane & 15) * ld + ((lane >> 4) << 3);
  v8h lo = *(const v8h*)(r);
  v8h hi = *(const v8h*)(r + 16);
  v16h b;
#pragma unroll
  for (int i = 0; i < 8; ++i) { b[i] = lo[i]; b[8 + i] = hi[i]; }
  return b;
}

// ---------------------------------------------------------------------------
// Problem constants: B=4, C=512, N=4096, Cq=64
#define XSTR 520   // f16 LDS row stride for 512-wide transposed tiles (16B aligned)
#define OST  513   // f32 LDS row stride for O transpose staging (conflict-free)

__global__ __launch_bounds__(256) void ca_cvt_weights(
    const float* __restrict__ Wq, const float* __restrict__ Wk,
    const float* __restrict__ Wv,
    _Float16* __restrict__ Wqh, _Float16* __restrict__ Wkh,
    _Float16* __restrict__ Wvh) {
  int i = blockIdx.x * 256 + threadIdx.x;
  if (i < 32768) { Wqh[i] = (_Float16)Wq[i]; Wkh[i] = (_Float16)Wk[i]; }
  if (i < 262144) Wvh[i] = (_Float16)Wv[i];
}

// Projections: Qh[b][n][cq], Kh[b][m][cq] (row-major, f16), Vt[b][c][m] (f16).
// One workgroup = one (b, 64-pixel tile). 256 threads = 8 waves.
__global__ __launch_bounds__(256) void ca_proj(
    const float* __restrict__ x, const float* __restrict__ skel,
    const float* __restrict__ bq, const float* __restrict__ bk,
    const float* __restrict__ bv,
    const _Float16* __restrict__ Wqh, const _Float16* __restrict__ Wkh,
    const _Float16* __restrict__ Wvh,
    _Float16* __restrict__ Qh, _Float16* __restrict__ Kh,
    _Float16* __restrict__ Vt) {
  extern __shared__ char smem[];
  _Float16* xt = (_Float16*)smem;        // [64][XSTR]  x^T tile, f16
  _Float16* st = xt + 64 * XSTR;         // [64][XSTR]  skel^T tile, f16
  const int tid = threadIdx.x;
  const int b = blockIdx.y;
  const int n0 = blockIdx.x * 64;

  // load + transpose + cvt to f16 (coalesced over n)
  for (int i = tid; i < 512 * 64; i += 256) {
    int c = i >> 6, n = i & 63;
    long g = ((long)(b * 512 + c) << 12) + n0 + n;
    xt[n * XSTR + c] = (_Float16)x[g];
    st[n * XSTR + c] = (_Float16)skel[g];
  }
  __syncthreads();

  const int w = tid >> 5, lane = tid & 31;
  const int hi = lane >> 4, col16 = lane & 15;

  // Q and K: 64x64 output each -> 16 tiles, 2 per wave
#pragma unroll
  for (int u = 0; u < 2; ++u) {
    int t = w + u * 8;
    int rt = t >> 2, ct = t & 3;
    v8f aq = {}; v8f ak = {};
    for (int ks = 0; ks < 16; ++ks) {
      v16h aA = frag_a(xt + rt * 16 * XSTR + ks * 32, XSTR, lane);
      v16h bQ = frag_bt(Wqh + (ct * 16) * 512 + ks * 32, 512, lane);
      aq = WMMA_F16(aA, bQ, aq);
      v16h aS = frag_a(st + rt * 16 * XSTR + ks * 32, XSTR, lane);
      v16h bK = frag_bt(Wkh + (ct * 16) * 512 + ks * 32, 512, lane);
      ak = WMMA_F16(aS, bK, ak);
    }
    int cq = ct * 16 + col16;
    float bqv = bq[cq], bkv = bk[cq];
#pragma unroll
    for (int i = 0; i < 8; ++i) {
      int row = rt * 16 + i + hi * 8;
      long base = ((long)((b << 12) + n0 + row) << 6) + cq;
      Qh[base] = (_Float16)(aq[i] + bqv);
      Kh[base] = (_Float16)(ak[i] + bkv);
    }
  }

  // V: 64x512 output -> 128 tiles, 16 per wave; store transposed Vt[c][m]
  for (int u = 0; u < 16; ++u) {
    int t = w * 16 + u;
    int rt = t & 3, ct = t >> 2;          // rt over m-tiles(4), ct over c-tiles(32)
    v8f acc = {};
    for (int ks = 0; ks < 16; ++ks) {
      v16h aS = frag_a(st + rt * 16 * XSTR + ks * 32, XSTR, lane);
      v16h bV = frag_bt(Wvh + (ct * 16) * 512 + ks * 32, 512, lane);
      acc = WMMA_F16(aS, bV, acc);
    }
    int c = ct * 16 + col16;
    float bvv = bv[c];
    v8h pk;
#pragma unroll
    for (int i = 0; i < 8; ++i) pk[i] = (_Float16)(acc[i] + bvv);
    // lane's 8 acc rows are m = rt*16 + hi*8 + 0..7 (contiguous): one 16B store
    *(v8h*)(Vt + ((long)(b * 512 + c) << 12) + n0 + rt * 16 + hi * 8) = pk;
  }
}

// Flash attention: one workgroup = (b, 64 query rows). 512 threads = 16 waves
// in a 4x4 grid: rt = wave>>2 (query rows), cw = wave&3 (key cols for S /
// 128 V columns for PV).
__global__ __launch_bounds__(512) void ca_attn(
    const float* __restrict__ x, const float* __restrict__ gamma,
    const _Float16* __restrict__ Qh, const _Float16* __restrict__ Kh,
    const _Float16* __restrict__ Vt, float* __restrict__ outp) {
  extern __shared__ char smem[];
  float* m_i     = (float*)smem;          // [64] running max
  float* l_i     = m_i + 64;              // [64] running sum
  float* scale   = l_i + 64;              // [64] exp(m_old-m_new)
  float* rowmaxA = scale + 64;            // [4][64] per-colwave row max
  float* rowsumA = rowmaxA + 256;         // [4][64] per-colwave row sum
  _Float16* pS   = (_Float16*)(rowsumA + 256); // [64][72] P block, f16
  float* obuf    = (float*)(pS + 64 * 72);     // [64][OST] O transpose staging

  const int tid = threadIdx.x;
  const int b = blockIdx.y;
  const int q0 = blockIdx.x * 64;
  const int w = tid >> 5, lane = tid & 31;
  const int rt = w >> 2, cw = w & 3;
  const int hi = lane >> 4, col16 = lane & 15;

  if (tid < 64) { m_i[tid] = -3.0e38f; l_i[tid] = 0.0f; }

  // Q A-fragments live in registers for the whole kernel
  const _Float16* qp = Qh + ((long)((b << 12) + q0 + rt * 16) << 6);
  v16h qa0 = frag_a(qp, 64, lane);
  v16h qa1 = frag_a(qp + 32, 64, lane);

  v8f o[8] = {};
  __syncthreads();

  for (int kb = 0; kb < 64; ++kb) {
    const int m0 = kb * 64;

    // --- S = Q K^T (one 16x16 tile per wave, k = 64) ---
    const _Float16* kp = Kh + ((long)((b << 12) + m0 + cw * 16) << 6);
    v16h kb0 = frag_bt(kp, 64, lane);
    v16h kb1 = frag_bt(kp + 32, 64, lane);
    v8f s = {};
    s = WMMA_F16(qa0, kb0, s);
    s = WMMA_F16(qa1, kb1, s);

    // --- per-row max across the 16 lanes of each half-wave ---
#pragma unroll
    for (int i = 0; i < 8; ++i) {
      float rm = s[i];
      rm = fmaxf(rm, __shfl_xor(rm, 1));
      rm = fmaxf(rm, __shfl_xor(rm, 2));
      rm = fmaxf(rm, __shfl_xor(rm, 4));
      rm = fmaxf(rm, __shfl_xor(rm, 8));
      if (col16 == 0) rowmaxA[cw * 64 + rt * 16 + i + hi * 8] = rm;
    }
    __syncthreads();

    if (tid < 64) {
      float mb = fmaxf(fmaxf(rowmaxA[tid], rowmaxA[64 + tid]),
                       fmaxf(rowmaxA[128 + tid], rowmaxA[192 + tid]));
      float mo = m_i[tid];
      float mn = fmaxf(mo, mb);
      m_i[tid] = mn;
      scale[tid] = __expf(mo - mn);
    }
    __syncthreads();

    float mn8[8], sc8[8];
#pragma unroll
    for (int i = 0; i < 8; ++i) {
      int rr = rt * 16 + i + hi * 8;
      mn8[i] = m_i[rr];
      sc8[i] = scale[rr];
    }
    // --- P = exp(S - m_new); share via LDS; per-row partial sums ---
#pragma unroll
    for (int i = 0; i < 8; ++i) {
      float p = __expf(s[i] - mn8[i]);
      pS[(rt * 16 + i + hi * 8) * 72 + cw * 16 + col16] = (_Float16)p;
      float rs = p;
      rs += __shfl_xor(rs, 1);
      rs += __shfl_xor(rs, 2);
      rs += __shfl_xor(rs, 4);
      rs += __shfl_xor(rs, 8);
      if (col16 == 0) rowsumA[cw * 64 + rt * 16 + i + hi * 8] = rs;
    }
    // --- online-softmax rescale of O accumulators ---
#pragma unroll
    for (int j = 0; j < 8; ++j)
#pragma unroll
      for (int i = 0; i < 8; ++i) o[j][i] *= sc8[i];
    __syncthreads();

    if (tid < 64) {
      l_i[tid] = l_i[tid] * scale[tid] + rowsumA[tid] + rowsumA[64 + tid] +
                 rowsumA[128 + tid] + rowsumA[192 + tid];
    }

    // --- O += P * V  (8 column tiles per wave, k = 64) ---
    v16h pa0 = frag_a(pS + rt * 16 * 72, 72, lane);
    v16h pa1 = frag_a(pS + rt * 16 * 72 + 32, 72, lane);
#pragma unroll
    for (int j = 0; j < 8; ++j) {
      const _Float16* vp =
          Vt + ((long)(b * 512 + cw * 128 + j * 16) << 12) + m0;
      v16h vb0 = frag_bt(vp, 4096, lane);       // L2-resident V^T stream
      v16h vb1 = frag_bt(vp + 32, 4096, lane);
      o[j] = WMMA_F16(pa0, vb0, o[j]);
      o[j] = WMMA_F16(pa1, vb1, o[j]);
    }
    __syncthreads();
  }

  // stage O in LDS (transpose) for coalesced global stores
#pragma unroll
  for (int j = 0; j < 8; ++j) {
    int c = cw * 128 + j * 16 + col16;
#pragma unroll
    for (int i = 0; i < 8; ++i)
      obuf[(rt * 16 + i + hi * 8) * OST + c] = o[j][i];
  }
  __syncthreads();

  const float g = gamma[0];
  for (int it = 0; it < 64; ++it) {
    int idx = it * 512 + tid;
    int c = idx >> 6, r = idx & 63;
    long gidx = ((long)(b * 512 + c) << 12) + q0 + r;
    float xv = x[gidx];
    outp[gidx] = g * (obuf[r * OST + c] / l_i[r]) + xv;  // out = gamma*O + x
    outp[8388608 + gidx] = xv;                           // tuple out #2: x flat
  }
}

extern "C" void kernel_launch(void* const* d_in, const int* in_sizes, int n_in,
                              void* d_out, int out_size, void* d_ws,
                              size_t ws_size, hipStream_t stream) {
  (void)in_sizes; (void)n_in; (void)out_size; (void)ws_size;
  const float* x     = (const float*)d_in[0];
  // d_in[1] = style (unused by reference math)
  const float* skel  = (const float*)d_in[2];
  const float* Wq    = (const float*)d_in[3];
  const float* bq    = (const float*)d_in[4];
  const float* Wk    = (const float*)d_in[5];
  const float* bk    = (const float*)d_in[6];
  const float* Wv    = (const float*)d_in[7];
  const float* bv    = (const float*)d_in[8];
  const float* gamma = (const float*)d_in[9];
  float* outp = (float*)d_out;

  // workspace layout (f16 halves): ~20.6 MB total
  _Float16* ws  = (_Float16*)d_ws;
  _Float16* Wqh = ws;                       // 64*512
  _Float16* Wkh = Wqh + 32768;              // 64*512
  _Float16* Wvh = Wkh + 32768;              // 512*512
  _Float16* Qh  = Wvh + 262144;             // 4*4096*64
  _Float16* Kh  = Qh + 4 * 4096 * 64;       // 4*4096*64
  _Float16* Vt  = Kh + 4 * 4096 * 64;       // 4*512*4096  (c-major!)

  ca_cvt_weights<<<1024, 256, 0, stream>>>(Wq, Wk, Wv, Wqh, Wkh, Wvh);

  dim3 g1(64, 4);
  size_t shm1 = (size_t)2 * 64 * XSTR * sizeof(_Float16);  // 133,120 B
  ca_proj<<<g1, 256, shm1, stream>>>(x, skel, bq, bk, bv, Wqh, Wkh, Wvh, Qh,
                                     Kh, Vt);

  dim3 g2(64, 4);
  size_t shm2 = 2816 + (size_t)64 * 72 * 2 + (size_t)64 * OST * 4;  // 143,360 B
  ca_attn<<<g2, 512, shm2, stream>>>(x, gamma, Qh, Kh, Vt, outp);
}
